// AttLayer_40303973106430
// MI455X (gfx1250) — matrix operand order repaired and checked by reading, not compiled, not run on hardware
//
#include <hip/hip_runtime.h>
#include <cstdint>
#include <cstddef>

// ---- problem constants (match reference) ----
#define B_   32
#define S_   2048
#define D_   1024
#define A_   256
#define EPS_ 1e-7f
#define KT_  (D_ / 32)   // 32 k-tiles of K=32
#define NT_  (A_ / 16)   // 16 n-tiles of N=16

typedef __bf16 v16bf __attribute__((ext_vector_type(16)));
typedef float  v8f   __attribute__((ext_vector_type(8)));

// fp32 pair -> packed bf16 dword (lo in [15:0], hi in [31:16]), round-to-nearest
#if __has_builtin(__builtin_amdgcn_cvt_pk_bf16_f32)
typedef __bf16 v2bf __attribute__((ext_vector_type(2)));
__device__ __forceinline__ unsigned int pack_bf16(float lo, float hi) {
    union { v2bf v; unsigned int u; } cv;
    cv.v = __builtin_amdgcn_cvt_pk_bf16_f32(lo, hi);     // 1x v_cvt_pk_bf16_f32
    return cv.u;
}
#elif __has_builtin(__builtin_amdgcn_perm)
__device__ __forceinline__ unsigned int pack_bf16(float lo, float hi) {
    unsigned int a = __float_as_uint(lo) + 0x8000u;      // round to nearest
    unsigned int b = __float_as_uint(hi) + 0x8000u;
    // v_perm_b32: {src0,src1} bytes; sel 0x07060302 -> [b.b3 b.b2 a.b3 a.b2]
    return __builtin_amdgcn_perm(b, a, 0x07060302u);     // 3 VALU total
}
#else
__device__ __forceinline__ unsigned int pack_bf16(float lo, float hi) {
    unsigned int a = __float_as_uint(lo);
    unsigned int b = __float_as_uint(hi);
    a = (a + 0x7FFFu + ((a >> 16) & 1u)) >> 16;
    b = (b + 0x7FFFu + ((b >> 16) & 1u)) >> 16;
    return (b << 16) | (a & 0xFFFFu);
}
#endif

// ---------------------------------------------------------------------------
// Kernel 1: pack w [D,A] fp32 row-major -> bf16 in WMMA B-fragment lane order.
// wp index = ((kt*NT_ + nt)*32 + lane)*8 + j  (dwords; 32B contiguous per lane)
// B k-order chosen here (16-bit, 32x16): lanes 0-15 hold K=0..15, lanes 16-31 hold K=16..31;
// dword j of a lane holds K = base+2j, base+2j+1 for column n = nt*16 + (lane&15).
// ---------------------------------------------------------------------------
__global__ __launch_bounds__(256) void pack_w_bf16(const float* __restrict__ w,
                                                   unsigned int* __restrict__ wp) {
    int idx  = blockIdx.x * 256 + threadIdx.x;       // 0 .. 131071
    int j    = idx & 7;
    int lane = (idx >> 3) & 31;
    int nt   = (idx >> 8) & 15;
    int kt   = idx >> 12;
    int n    = nt * 16 + (lane & 15);
    int k0   = kt * 32 + ((lane >> 4) << 4) + 2 * j;
    float lo = w[(size_t)k0 * A_ + n];
    float hi = w[(size_t)(k0 + 1) * A_ + n];
    wp[idx] = pack_bf16(lo, hi);
}

// ---------------------------------------------------------------------------
// Kernel 2: additive-attention scores.
// Block = 256 threads = 8 waves; each wave owns a 16-row M-tile (rows of b*s).
// Per k-tile: stage 16KB of packed w into LDS, build bf16 A-frag from global x,
// run 16 x v_wmma_f32_16x16x32_bf16 (one per N-tile) into resident f32 C tiles.
// Epilogue: score = sum_n u[n]*tanh(C + bias[n]); e = exp(score)*mask.
// ---------------------------------------------------------------------------
__global__ __launch_bounds__(256) void scores_kernel(
    const float* __restrict__ x, const unsigned char* __restrict__ mask,
    const unsigned int* __restrict__ wp, const float* __restrict__ bias,
    const float* __restrict__ u, float* __restrict__ e_out,
    float* __restrict__ esum_part) {
    __shared__ unsigned int lbw[4096];   // one k-tile of packed w: 16 KB
    __shared__ float sE;

    const int tid  = threadIdx.x;
    const int wave = tid >> 5;
    const int lane = tid & 31;
    const int lo16 = lane & 15;
    const int hi   = lane >> 4;
    const int r0   = blockIdx.x * 128 + wave * 16;   // first global row of tile

    if (tid == 0) sE = 0.0f;

    v8f c[NT_];
    const v8f vzero = {0.f, 0.f, 0.f, 0.f, 0.f, 0.f, 0.f, 0.f};
#pragma unroll
    for (int nt = 0; nt < NT_; ++nt) c[nt] = vzero;

    // A k-order chosen here (16-bit, 16x32, same as B): lane<16 -> row M=lane, K 0..15;
    // lane>=16 -> row M=lane-16, K 16..31.  => per-lane base = hi*16.
    const float* xrow = x + (size_t)(r0 + lo16) * D_ + hi * 16;

    for (int kt = 0; kt < KT_; ++kt) {
        __syncthreads();   // previous iteration done reading LDS
        {
            const uint4* src = (const uint4*)(wp + (size_t)kt * 4096);
            uint4* dst = (uint4*)lbw;
#pragma unroll
            for (int t = 0; t < 4; ++t)
                dst[tid + t * 256] = src[tid + t * 256];
        }
        __syncthreads();

        // prefetch next k-step's x cachelines (global_prefetch_b8)
        if (kt + 1 < KT_)
            __builtin_prefetch(xrow + (kt + 1) * 32, 0, 3);

        // build bf16 A fragment from fp32 x (two contiguous 32B loads per lane)
        union { unsigned int u[8]; v16bf v; } afrag;
        {
            const float4* p0 = (const float4*)(xrow + kt * 32);
            const float4* p1 = (const float4*)(xrow + kt * 32 + 8);
            float4 f0 = p0[0], f1 = p0[1];
            float4 g0 = p1[0], g1 = p1[1];
            afrag.u[0] = pack_bf16(f0.x, f0.y);
            afrag.u[1] = pack_bf16(f0.z, f0.w);
            afrag.u[2] = pack_bf16(f1.x, f1.y);
            afrag.u[3] = pack_bf16(f1.z, f1.w);
            afrag.u[4] = pack_bf16(g0.x, g0.y);
            afrag.u[5] = pack_bf16(g0.z, g0.w);
            afrag.u[6] = pack_bf16(g1.x, g1.y);
            afrag.u[7] = pack_bf16(g1.z, g1.w);
        }

#pragma unroll
        for (int nt = 0; nt < NT_; ++nt) {
            union { unsigned int u[8]; v16bf v; } bfrag;
            const unsigned int* bl = &lbw[(nt * 32 + lane) * 8];
#pragma unroll
            for (int j = 0; j < 8; ++j) bfrag.u[j] = bl[j];
            c[nt] = __builtin_amdgcn_wmma_f32_16x16x32_bf16(
                false, afrag.v, false, bfrag.v, (short)0, c[nt], false, false);
        }
    }

    // Epilogue. C layout: lane gives N = lo16 (per n-tile), VGPR r gives M = r + 8*hi.
    float p[8];
#pragma unroll
    for (int r = 0; r < 8; ++r) p[r] = 0.0f;
#pragma unroll
    for (int nt = 0; nt < NT_; ++nt) {
        int n = nt * 16 + lo16;
        float bn = bias[n];
        float un = u[n];
#pragma unroll
        for (int r = 0; r < 8; ++r)
            p[r] += tanhf(c[nt][r] + bn) * un;
    }
    // butterfly-reduce over the 16 lanes of each half (sums over all 256 n)
#pragma unroll
    for (int m = 1; m < 16; m <<= 1) {
#pragma unroll
        for (int r = 0; r < 8; ++r)
            p[r] += __shfl_xor(p[r], m, 32);
    }
    if (lo16 == 0) {
        float es = 0.0f;
#pragma unroll
        for (int r = 0; r < 8; ++r) {
            int g = r0 + hi * 8 + r;               // flat (b*S + s)
            float e = __expf(p[r]);
            e *= mask[g] ? 1.0f : 0.0f;
            e_out[g] = e;
            es += e;
        }
        atomicAdd(&sE, es);                        // ds_add_f32
    }
    __syncthreads();
    if (tid == 0) esum_part[blockIdx.x] = sE;      // all 128 rows share one batch
}

// ---------------------------------------------------------------------------
// Kernel 3: partial weighted pooling. Grid 512 = 32 batches x 16 s-chunks.
// Each thread owns 4 columns (float4); coalesced b128 streaming of x.
// ---------------------------------------------------------------------------
__global__ __launch_bounds__(256) void pool_partial(const float* __restrict__ x,
                                                    const float* __restrict__ e_ws,
                                                    float* __restrict__ part) {
    const int b     = blockIdx.x >> 4;
    const int chunk = blockIdx.x & 15;
    const int d4    = threadIdx.x * 4;
    const int s0    = chunk * (S_ / 16);           // 128 rows per chunk
    const float* xb = x + ((size_t)b * S_ + s0) * D_;
    const float* eb = e_ws + (size_t)b * S_ + s0;
    float4 acc = {0.f, 0.f, 0.f, 0.f};
    for (int s = 0; s < S_ / 16; ++s) {
        float e = eb[s];
        float4 xv = *(const float4*)(xb + (size_t)s * D_ + d4);
        acc.x += e * xv.x;
        acc.y += e * xv.y;
        acc.z += e * xv.z;
        acc.w += e * xv.w;
    }
    *(float4*)(part + (size_t)blockIdx.x * D_ + d4) = acc;
}

// ---------------------------------------------------------------------------
// Kernel 4: reduce 16 partials per batch and normalize by (sum(e) + EPS).
// ---------------------------------------------------------------------------
__global__ __launch_bounds__(256) void finalize(const float* __restrict__ part,
                                                const float* __restrict__ esum_part,
                                                float* __restrict__ out) {
    const int b  = blockIdx.x;
    const int d4 = threadIdx.x * 4;
    float se = 0.0f;
#pragma unroll
    for (int c = 0; c < 16; ++c) se += esum_part[b * 16 + c];
    const float inv = 1.0f / (se + EPS_);
    float4 acc = {0.f, 0.f, 0.f, 0.f};
    for (int c = 0; c < 16; ++c) {
        float4 v = *(const float4*)(part + ((size_t)(b * 16 + c)) * D_ + d4);
        acc.x += v.x; acc.y += v.y; acc.z += v.z; acc.w += v.w;
    }
    acc.x *= inv; acc.y *= inv; acc.z *= inv; acc.w *= inv;
    *(float4*)(out + (size_t)b * D_ + d4) = acc;
}

// ---------------------------------------------------------------------------
// Launch. Inputs: x[32,2048,1024] f32, mask[32,2048] bool(u8),
//                 w[1024,256] f32, b[256] f32, u[256,1] f32. Out: [32,1024] f32.
// Workspace layout (bytes): wp 512K | e 256K | esum 2K(+pad) | part 2M
// ---------------------------------------------------------------------------
extern "C" void kernel_launch(void* const* d_in, const int* in_sizes, int n_in,
                              void* d_out, int out_size, void* d_ws, size_t ws_size,
                              hipStream_t stream) {
    (void)in_sizes; (void)n_in; (void)out_size; (void)ws_size;
    const float*         x    = (const float*)d_in[0];
    const unsigned char* mask = (const unsigned char*)d_in[1];
    const float*         w    = (const float*)d_in[2];
    const float*         bias = (const float*)d_in[3];
    const float*         u    = (const float*)d_in[4];
    float* out = (float*)d_out;

    char* ws = (char*)d_ws;
    unsigned int* wp   = (unsigned int*)(ws);               // 131072 dwords = 512 KB
    float*        e_ws = (float*)(ws + (512 << 10));        // 65536 f = 256 KB
    float*        esum = (float*)(ws + (768 << 10));        // 512 f
    float*        part = (float*)(ws + (772 << 10));        // 512*1024 f = 2 MB

    pack_w_bf16  <<<512, 256, 0, stream>>>(w, wp);
    scores_kernel<<<512, 256, 0, stream>>>(x, mask, wp, bias, u, e_ws, esum);
    pool_partial <<<512, 256, 0, stream>>>(x, e_ws, part);
    finalize     <<<32,  256, 0, stream>>>(part, esum, out);
}
